// BiLSTMClassifier_6734508720648
// MI455X (gfx1250) — matrix operand order repaired and checked by reading, not compiled, not run on hardware
//
#include <hip/hip_runtime.h>
#include <cmath>

// ---------------------------------------------------------------------------
// BiLSTM classifier for MI455X (gfx1250, wave32, WMMA).
// bf16 WMMA (f32 accum) for all matmuls. Input projections are parallel
// GEMMs that emit results in WMMA *fragment layout* so the recurrent scan
// can double-buffer them with b128 loads straight into C accumulators.
// Scan: 8 batch-tiles x 2 directions; W_hh lives in VGPRs across all T steps.
// ---------------------------------------------------------------------------

typedef unsigned short u16;
typedef __attribute__((ext_vector_type(16))) __bf16 v16bf;
typedef __attribute__((ext_vector_type(8)))  float  v8f;

#define B_  128
#define T_  1024
#define E_  100
#define H_  128
#define G_  512      // 4*H gates
#define C_  2
#define NT_ 32       // G_/16 gate n-tiles
#define KP0 128      // E padded to multiple of 32
#define KP1 256      // 2*H
#define HP  136      // padded bf16 LDS row stride for h   (136*2B = 4.25 banks)
#define GP  516      // padded f32 LDS row stride for gates (516%64 = 4 banks)

__device__ __forceinline__ u16 f2bf(float f) {           // round-to-nearest-even
  unsigned int u = __float_as_uint(f);
  u += 0x7FFFu + ((u >> 16) & 1u);
  return (u16)(u >> 16);
}
// sigmoid via tanh: one transcendental instead of exp+rcp
__device__ __forceinline__ float sigm(float x) { return 0.5f * tanhf(0.5f * x) + 0.5f; }

// Build a 16x__bf16 fragment from two 16-byte chunks `gap` elements apart.
// A-frag (16x32, MxK): lane holds K=[kb..kb+7] and [kb+16..kb+23] -> gap=16.
// B-frag (32x16, KxN): lane holds the same K split for one column   -> gap=16.
__device__ __forceinline__ v16bf ld_frag(const u16* p, int gap) {
  union { uint4 u[2]; v16bf v; } t;
  t.u[0] = *(const uint4*)(p);
  t.u[1] = *(const uint4*)(p + gap);
  return t.v;
}
__device__ __forceinline__ v8f wmma_bf16(v16bf a, v16bf b, v8f c) {
  return __builtin_amdgcn_wmma_f32_16x16x32_bf16(false, a, false, b, (short)0, c,
                                                 false, false);
}
union FragF { uint4 u[2]; v8f v; };

// ---------------------------------------------------------------------------
// Weight prep: f32 -> bf16 (pad E 100->128), fold b_ih + b_hh.
// grid = G_, block = 256.
// ---------------------------------------------------------------------------
__global__ void prep_weights(
    const float* __restrict__ wih0f, const float* __restrict__ whh0f,
    const float* __restrict__ bih0f, const float* __restrict__ bhh0f,
    const float* __restrict__ wih0b, const float* __restrict__ whh0b,
    const float* __restrict__ bih0b, const float* __restrict__ bhh0b,
    const float* __restrict__ wih1f, const float* __restrict__ whh1f,
    const float* __restrict__ bih1f, const float* __restrict__ bhh1f,
    const float* __restrict__ wih1b, const float* __restrict__ whh1b,
    const float* __restrict__ bih1b, const float* __restrict__ bhh1b,
    u16* __restrict__ dwih0f, u16* __restrict__ dwih0b,
    u16* __restrict__ dwhh0f, u16* __restrict__ dwhh0b,
    u16* __restrict__ dwih1f, u16* __restrict__ dwih1b,
    u16* __restrict__ dwhh1f, u16* __restrict__ dwhh1b,
    float* __restrict__ db0f, float* __restrict__ db0b,
    float* __restrict__ db1f, float* __restrict__ db1b)
{
  const int r = blockIdx.x;          // gate row 0..511
  const int tid = threadIdx.x;
  for (int k = tid; k < KP0; k += blockDim.x) {
    dwih0f[r*KP0 + k] = (k < E_) ? f2bf(wih0f[r*E_ + k]) : (u16)0;
    dwih0b[r*KP0 + k] = (k < E_) ? f2bf(wih0b[r*E_ + k]) : (u16)0;
    dwhh0f[r*H_ + k]  = f2bf(whh0f[r*H_ + k]);
    dwhh0b[r*H_ + k]  = f2bf(whh0b[r*H_ + k]);
    dwhh1f[r*H_ + k]  = f2bf(whh1f[r*H_ + k]);
    dwhh1b[r*H_ + k]  = f2bf(whh1b[r*H_ + k]);
  }
  for (int k = tid; k < KP1; k += blockDim.x) {
    dwih1f[r*KP1 + k] = f2bf(wih1f[r*KP1 + k]);
    dwih1b[r*KP1 + k] = f2bf(wih1b[r*KP1 + k]);
  }
  if (tid == 0) {
    db0f[r] = bih0f[r] + bhh0f[r];
    db0b[r] = bih0b[r] + bhh0b[r];
    db1f[r] = bih1f[r] + bhh1f[r];
    db1b[r] = bih1b[r] + bhh1b[r];
  }
}

// ---------------------------------------------------------------------------
// Embedding gather -> bf16, K padded to 128. xe layout: [T*B][KP0], row=t*B+b.
// grid = T*B, block = 128.
// ---------------------------------------------------------------------------
__global__ void embed_pad(const int* __restrict__ x, const float* __restrict__ emb,
                          u16* __restrict__ xe)
{
  const int r = blockIdx.x;          // r = t*B + b
  const int b = r & (B_ - 1);
  const int t = r >> 7;              // /128
  const int e = threadIdx.x;
  const int tok = x[b*T_ + t];
  xe[(size_t)r*KP0 + e] = (e < E_) ? f2bf(emb[(size_t)tok*E_ + e]) : (u16)0;
}

// ---------------------------------------------------------------------------
// xp[dir] = A @ W[dir]^T + bias[dir] stored in WMMA FRAGMENT layout:
//   xp[((mtile*NT_ + nt)*32 + lane)*8 + r]
// where mtile = t*8 + bt (16 consecutive rows of [T*B]).
// A:[T*B][Kp] bf16, W:[512][Kp] bf16.
// grid = (T*B/16, 2 dirs), block = 256 (8 waves x 4 n-tiles).
// ---------------------------------------------------------------------------
__global__ void gemm_xp(const u16* __restrict__ A, int Kp,
                        const u16* __restrict__ W0, const u16* __restrict__ W1,
                        const float* __restrict__ bias0, const float* __restrict__ bias1,
                        float* __restrict__ out0, float* __restrict__ out1)
{
  const int dir = blockIdx.y;
  const u16* W = dir ? W1 : W0;
  const float* bias = dir ? bias1 : bias0;
  float* out = dir ? out1 : out0;
  const int mtile = blockIdx.x;
  const int tid = threadIdx.x, wave = tid >> 5, lane = tid & 31;
  const int ln = lane & 15, lh = lane >> 4;

  v8f acc[4];
#pragma unroll
  for (int j = 0; j < 4; ++j) {
    const int n = (wave*4 + j)*16 + ln;
    const float bv = bias[n];
#pragma unroll
    for (int r = 0; r < 8; ++r) acc[j][r] = bv;
  }
  const int kiters = Kp >> 5;
  const u16* arow = A + (size_t)(mtile*16 + ln) * Kp;
  for (int kk = 0; kk < kiters; ++kk) {
    const v16bf af = ld_frag(arow + kk*32 + lh*8, 16);
#pragma unroll
    for (int j = 0; j < 4; ++j) {
      const int n = (wave*4 + j)*16 + ln;
      const v16bf wfr = ld_frag(W + (size_t)n*Kp + kk*32 + lh*8, 16);
      acc[j] = wmma_bf16(af, wfr, acc[j]);
    }
  }
#pragma unroll
  for (int j = 0; j < 4; ++j) {
    const int nt = wave*4 + j;
    float* op = out + (((size_t)mtile*NT_ + nt)*32 + lane)*8;
    FragF fr; fr.v = acc[j];
    *(uint4*)(op)     = fr.u[0];
    *(uint4*)(op + 4) = fr.u[1];
  }
}

// ---------------------------------------------------------------------------
// Recurrent scan. One workgroup per (16-row batch tile, direction); 512 thr =
// 16 waves, each wave owns 2 of the 32 gate n-tiles with W_hh B-fragments
// resident in VGPRs. xp arrives in fragment layout and is double-buffered
// (next step's fragments are in flight while this step computes).
// grid = (B/16, 2), block = 512.
// ---------------------------------------------------------------------------
__global__ void lstm_scan(const float* __restrict__ xp0, const float* __restrict__ xp1,
                          const u16* __restrict__ whh0, const u16* __restrict__ whh1,
                          u16* __restrict__ y_out,     // [T*B][2H] bf16 or null
                          float* __restrict__ hT_out)  // [B][2H] f32 or null
{
  const int dir = blockIdx.y;
  const float* xp = dir ? xp1 : xp0;
  const u16* whh = dir ? whh1 : whh0;
  const int coloff = dir * H_;
  const int bt = blockIdx.x;
  const int tid = threadIdx.x, wave = tid >> 5, lane = tid & 31;
  const int ln = lane & 15, lh = lane >> 4;

  __shared__ __align__(16) u16 h_lds[16 * HP];
  __shared__ __align__(16) float c_lds[16 * H_];
  __shared__ __align__(16) float gates[16 * GP];

  // Preload W_hh B-fragments into registers (2 ntiles x 4 k-steps x 8 VGPRs).
  v16bf wf[2][4];
#pragma unroll
  for (int j = 0; j < 2; ++j) {
    const int n = (wave*2 + j)*16 + ln;
#pragma unroll
    for (int kk = 0; kk < 4; ++kk)
      wf[j][kk] = ld_frag(whh + (size_t)n*H_ + kk*32 + lh*8, 16);
  }
  for (int e = tid; e < 16*HP; e += blockDim.x) h_lds[e] = 0;
  for (int e = tid; e < 16*H_; e += blockDim.x) c_lds[e] = 0.0f;
  __syncthreads();

  // fragment address for timestep t and this wave's j-th n-tile
  auto fragp = [&](int t, int j) -> const float* {
    const int mt = t*8 + bt;
    const int nt = wave*2 + j;
    return xp + (((size_t)mt*NT_ + nt)*32 + lane)*8;
  };

  // prime the xp double buffer with step 0
  FragF cur[2];
  {
    const int t0 = dir ? (T_ - 1) : 0;
#pragma unroll
    for (int j = 0; j < 2; ++j) {
      const float* p = fragp(t0, j);
      cur[j].u[0] = *(const uint4*)(p);
      cur[j].u[1] = *(const uint4*)(p + 4);
    }
  }

  for (int it = 0; it < T_; ++it) {
    const int tt = dir ? (T_ - 1 - it) : it;
    // kick off next step's xp fragments (clamped; latency hides under step)
    const int itn = (it + 1 < T_) ? (it + 1) : it;
    const int ttn = dir ? (T_ - 1 - itn) : itn;
    FragF nxt[2];
#pragma unroll
    for (int j = 0; j < 2; ++j) {
      const float* p = fragp(ttn, j);
      nxt[j].u[0] = *(const uint4*)(p);
      nxt[j].u[1] = *(const uint4*)(p + 4);
    }

    // A fragments from current hidden state (shared across this wave's ntiles)
    v16bf af[4];
#pragma unroll
    for (int kk = 0; kk < 4; ++kk)
      af[kk] = ld_frag(&h_lds[ln*HP + kk*32 + lh*8], 16);

#pragma unroll
    for (int j = 0; j < 2; ++j) {
      const int n = (wave*2 + j)*16 + ln;
      v8f a = cur[j].v;                                  // C init = xp + bias
#pragma unroll
      for (int kk = 0; kk < 4; ++kk) a = wmma_bf16(af[kk], wf[j][kk], a);
#pragma unroll
      for (int r = 0; r < 8; ++r) gates[(lh*8 + r)*GP + n] = a[r];
    }
    __syncthreads();

    // Elementwise cell update: 16x128 states, 4 consecutive j per thread.
    {
      const int e0 = tid * 4;
      const int m = e0 >> 7, j0 = e0 & (H_ - 1);
      const float4 gi = *(const float4*)&gates[m*GP + j0];
      const float4 gf = *(const float4*)&gates[m*GP + j0 +   H_];
      const float4 gg = *(const float4*)&gates[m*GP + j0 + 2*H_];
      const float4 go = *(const float4*)&gates[m*GP + j0 + 3*H_];
      float4 c4 = *(const float4*)&c_lds[m*H_ + j0];
      float h0, h1, h2, h3;
      c4.x = sigm(gf.x)*c4.x + sigm(gi.x)*tanhf(gg.x); h0 = sigm(go.x)*tanhf(c4.x);
      c4.y = sigm(gf.y)*c4.y + sigm(gi.y)*tanhf(gg.y); h1 = sigm(go.y)*tanhf(c4.y);
      c4.z = sigm(gf.z)*c4.z + sigm(gi.z)*tanhf(gg.z); h2 = sigm(go.z)*tanhf(c4.z);
      c4.w = sigm(gf.w)*c4.w + sigm(gi.w)*tanhf(gg.w); h3 = sigm(go.w)*tanhf(c4.w);
      *(float4*)&c_lds[m*H_ + j0] = c4;
      ushort4 hb = make_ushort4(f2bf(h0), f2bf(h1), f2bf(h2), f2bf(h3));
      *(ushort4*)&h_lds[m*HP + j0] = hb;
      if (y_out)
        *(ushort4*)&y_out[((size_t)tt*B_ + bt*16 + m)*(2*H_) + coloff + j0] = hb;
      if (hT_out && it == T_ - 1)
        *(float4*)&hT_out[(size_t)(bt*16 + m)*(2*H_) + coloff + j0] =
            make_float4(h0, h1, h2, h3);
    }
    __syncthreads();

    cur[0] = nxt[0];
    cur[1] = nxt[1];
  }
}

// ---------------------------------------------------------------------------
// Final FC: out[b][c] = combined[b] . fc_w[c] + fc_b[c].  1 block x 256 thr.
// ---------------------------------------------------------------------------
__global__ void fc_head(const float* __restrict__ combined,
                        const float* __restrict__ fc_w,
                        const float* __restrict__ fc_b,
                        float* __restrict__ out)
{
  const int tid = threadIdx.x;       // 256 = B*C
  const int b = tid >> 1, c = tid & 1;
  float s = fc_b[c];
  for (int k = 0; k < 2*H_; ++k)
    s += combined[b*2*H_ + k] * fc_w[c*2*H_ + k];
  out[b*C_ + c] = s;
}

// ---------------------------------------------------------------------------
extern "C" void kernel_launch(void* const* d_in, const int* in_sizes, int n_in,
                              void* d_out, int out_size, void* d_ws, size_t ws_size,
                              hipStream_t stream)
{
  const int*   x     = (const int*)  d_in[0];
  const float* emb   = (const float*)d_in[1];
  const float* wih0f = (const float*)d_in[2],  *whh0f = (const float*)d_in[3];
  const float* bih0f = (const float*)d_in[4],  *bhh0f = (const float*)d_in[5];
  const float* wih0b = (const float*)d_in[6],  *whh0b = (const float*)d_in[7];
  const float* bih0b = (const float*)d_in[8],  *bhh0b = (const float*)d_in[9];
  const float* wih1f = (const float*)d_in[10], *whh1f = (const float*)d_in[11];
  const float* bih1f = (const float*)d_in[12], *bhh1f = (const float*)d_in[13];
  const float* wih1b = (const float*)d_in[14], *whh1b = (const float*)d_in[15];
  const float* bih1b = (const float*)d_in[16], *bhh1b = (const float*)d_in[17];
  const float* fcw   = (const float*)d_in[18];
  const float* fcb   = (const float*)d_in[19];
  (void)in_sizes; (void)n_in; (void)out_size; (void)ws_size;

  // Workspace carve-out (~640 MB; xp buffers reused between layers).
  char* ws = (char*)d_ws; size_t o = 0;
  auto alloc = [&](size_t bytes) -> void* {
    void* p = ws + o; o = (o + bytes + 255) & ~(size_t)255; return p;
  };
  u16*   xe     = (u16*)  alloc((size_t)T_*B_*KP0*sizeof(u16));   //  32 MB
  u16*   y0     = (u16*)  alloc((size_t)T_*B_*2*H_*sizeof(u16));  //  64 MB
  float* xp0    = (float*)alloc((size_t)T_*B_*G_*sizeof(float));  // 256 MB
  float* xp1    = (float*)alloc((size_t)T_*B_*G_*sizeof(float));  // 256 MB
  u16*   dwih0f = (u16*)  alloc((size_t)G_*KP0*sizeof(u16));
  u16*   dwih0b = (u16*)  alloc((size_t)G_*KP0*sizeof(u16));
  u16*   dwhh0f = (u16*)  alloc((size_t)G_*H_*sizeof(u16));
  u16*   dwhh0b = (u16*)  alloc((size_t)G_*H_*sizeof(u16));
  u16*   dwih1f = (u16*)  alloc((size_t)G_*KP1*sizeof(u16));
  u16*   dwih1b = (u16*)  alloc((size_t)G_*KP1*sizeof(u16));
  u16*   dwhh1f = (u16*)  alloc((size_t)G_*H_*sizeof(u16));
  u16*   dwhh1b = (u16*)  alloc((size_t)G_*H_*sizeof(u16));
  float* db0f   = (float*)alloc(G_*sizeof(float));
  float* db0b   = (float*)alloc(G_*sizeof(float));
  float* db1f   = (float*)alloc(G_*sizeof(float));
  float* db1b   = (float*)alloc(G_*sizeof(float));
  float* comb   = (float*)alloc((size_t)B_*2*H_*sizeof(float));

  prep_weights<<<G_, 256, 0, stream>>>(
      wih0f, whh0f, bih0f, bhh0f, wih0b, whh0b, bih0b, bhh0b,
      wih1f, whh1f, bih1f, bhh1f, wih1b, whh1b, bih1b, bhh1b,
      dwih0f, dwih0b, dwhh0f, dwhh0b, dwih1f, dwih1b, dwhh1f, dwhh1b,
      db0f, db0b, db1f, db1b);

  embed_pad<<<T_*B_, KP0, 0, stream>>>(x, emb, xe);

  gemm_xp<<<dim3(T_*B_/16, 2), 256, 0, stream>>>(
      xe, KP0, dwih0f, dwih0b, db0f, db0b, xp0, xp1);

  lstm_scan<<<dim3(B_/16, 2), 512, 0, stream>>>(
      xp0, xp1, dwhh0f, dwhh0b, y0, nullptr);

  gemm_xp<<<dim3(T_*B_/16, 2), 256, 0, stream>>>(
      y0, KP1, dwih1f, dwih1b, db1f, db1b, xp0, xp1);

  lstm_scan<<<dim3(B_/16, 2), 512, 0, stream>>>(
      xp0, xp1, dwhh1f, dwhh1b, nullptr, comb);

  fc_head<<<1, B_*C_, 0, stream>>>(comb, fcw, fcb, (float*)d_out);
}
